// SPAttention_12249246728420
// MI455X (gfx1250) — matrix operand order repaired and checked by reading, not compiled, not run on hardware
//
#include <hip/hip_runtime.h>
#include <math.h>

// ---------- CDNA5 WMMA types ----------
typedef __attribute__((ext_vector_type(16))) __bf16         v16bf;
typedef __attribute__((ext_vector_type(8)))  float          v8f;
typedef __attribute__((ext_vector_type(8)))  unsigned short u16x8;

union Frag {                    // 16 bf16 = one WMMA A/B operand (8 VGPRs)
    struct { u16x8 lo, hi; } h;
    v16bf b;
};
union F8 { v8f v; float f[8]; };  // one 16x16 f32 C/D tile slice per lane

__device__ __forceinline__ unsigned short f2bf(float x) {
    union { float f; unsigned int u; } c; c.f = x;
    unsigned int u = c.u;
    unsigned int r = u + 0x7FFFu + ((u >> 16) & 1u);   // round-to-nearest-even
    return (unsigned short)(r >> 16);
}

__device__ __forceinline__ v16bf ldfrag(const unsigned short* p_lo,
                                        const unsigned short* p_hi) {
    Frag f;
    f.h.lo = *(const u16x8*)p_lo;   // 16-byte LDS load
    f.h.hi = *(const u16x8*)p_hi;   // 16-byte LDS load
    return f.b;
}

// ---------- gfx1250 async global->LDS copy (ASYNCcnt path), with fallback ----
#if __has_builtin(__builtin_amdgcn_global_load_async_to_lds_b128)
#define ASYNC_LDS 1
typedef int i32x4_vs __attribute__((vector_size(16)));  // matches builtin param
#endif

__device__ __forceinline__ void cp16(void* l, const void* g) {
#ifdef ASYNC_LDS
    __builtin_amdgcn_global_load_async_to_lds_b128(
        (__attribute__((address_space(1))) i32x4_vs*)(unsigned long long)(size_t)g,
        (__attribute__((address_space(3))) i32x4_vs*)(unsigned)(size_t)l, 0, 0);
#else
    *(uint4*)l = *(const uint4*)g;
#endif
}

__device__ __forceinline__ void cp_fence() {
#ifdef ASYNC_LDS
#if __has_builtin(__builtin_amdgcn_s_wait_asynccnt)
    __builtin_amdgcn_s_wait_asynccnt(0);
#else
    asm volatile("s_wait_asynccnt 0x0" ::: "memory");
#endif
#endif
}

// ---------- gfx1250 Tensor Data Mover (TENSORcnt path), with fallback -------
#if __has_builtin(__builtin_amdgcn_tensor_load_to_lds)
#define HAVE_TDM 1
typedef __attribute__((ext_vector_type(4))) unsigned int v4u;
typedef __attribute__((ext_vector_type(8))) int          v8i;
typedef __attribute__((ext_vector_type(4))) int          v4i;

// DMA a 64x64 bf16 tile (row stride `stride` elems) into LDS laid out as
// [64][64+8] ushorts: TDM pad feature = 4 DWORDs pad every 32 DWORDs (one row).
__device__ __forceinline__ void tdm_load_64x64_bf16(unsigned lds_addr,
                                                    const unsigned short* gsrc,
                                                    unsigned stride) {
    unsigned long long ga = (unsigned long long)(size_t)gsrc;
    v4u g0;
    g0.x = 1u;                                     // count=1 (valid), user mode
    g0.y = lds_addr;                               // lds_addr [63:32]
    g0.z = (unsigned)(ga & 0xFFFFFFFFu);           // global_addr [95:64]
    g0.w = (unsigned)((ga >> 32) & 0x1FFFFFFu)     // global_addr [120:96]
         | (2u << 30);                             // type=2 ("image")
    v8i g1 = {0, 0, 0, 0, 0, 0, 0, 0};
    g1[0] = (int)((1u << 16)       // data_size = 2 bytes
                | (1u << 20)       // pad_enable
                | (4u << 22)       // pad_interval: 2^4 * 8B = 32 DWORDs
                | (3u << 25));     // pad_amount: 4 DWORDs (= 8 bf16 pad)
    g1[1] = (int)((64u & 0xFFFFu) << 16);    // tensor_dim0[15:0] @ bits 63:48
    g1[2] = (int)((2048u & 0xFFFFu) << 16);  // tensor_dim1[15:0] @ bits 95:80
    g1[3] = (int)(64u << 16);                // tile_dim0 @ bits 127:112
    g1[4] = (int)(64u);                      // tile_dim1 @ bits 143:128
    g1[5] = (int)stride;                     // tensor_dim0_stride low 32
    v4i g2 = {0, 0, 0, 0};
    v4i g3 = {0, 0, 0, 0};
#if __clang_major__ >= 23
    v8i g4 = {0, 0, 0, 0, 0, 0, 0, 0};
    __builtin_amdgcn_tensor_load_to_lds(g0, g1, g2, g3, g4, 0);
#else
    __builtin_amdgcn_tensor_load_to_lds(g0, g1, g2, g3, 0);
#endif
}
#endif

// ---------- fp32 -> bf16 convert ----------
__global__ void f32_to_bf16_kernel(const float* __restrict__ in,
                                   unsigned short* __restrict__ out, int n) {
    int i = (blockIdx.x * blockDim.x + threadIdx.x) * 4;
    if (i < n) {
        float4 v = *(const float4*)(in + i);
        uint2 w;
        w.x = (unsigned)f2bf(v.x) | ((unsigned)f2bf(v.y) << 16);
        w.y = (unsigned)f2bf(v.z) | ((unsigned)f2bf(v.w) << 16);
        *(uint2*)(out + i) = w;
    }
}

// ---------- tiled bf16 WMMA GEMM:  C[M,N] = A[M,K] * B[K,N] + bias ----------
// 256 threads = 8 waves; block tile 128x128, K-step 32 (one wmma K).
// Wave (wm in 0..1, wn in 0..3) owns a 64x32 tile = 4x2 accumulators.
__device__ __forceinline__ void store_out(float* p, float v) { *p = v; }
__device__ __forceinline__ void store_out(unsigned short* p, float v) { *p = f2bf(v); }

template <typename OutT>
__global__ __launch_bounds__(256)
void gemm_bf16_wmma(const unsigned short* __restrict__ A,
                    const unsigned short* __restrict__ B,
                    const float* __restrict__ bias,
                    OutT* __restrict__ C,
                    int M, int N, int K) {
    constexpr int BM = 128, BN = 128, BK = 32, PAD = 8;
    __shared__ unsigned short As[BM][BK + PAD];     // row-major [m][k]
    __shared__ unsigned short Bt[BN][BK + PAD];     // transposed [n][k]

    const int tid  = threadIdx.x;
    const int bm   = blockIdx.y * BM, bn = blockIdx.x * BN;
    const int wave = tid >> 5, lane = tid & 31;
    const int wm   = wave >> 2, wn = wave & 3;
    const int lm   = lane & 15, lg = lane >> 4;

    F8 acc[4][2];
#pragma unroll
    for (int mi = 0; mi < 4; ++mi)
#pragma unroll
        for (int ni = 0; ni < 2; ++ni)
#pragma unroll
            for (int i = 0; i < 8; ++i) acc[mi][ni].f[i] = 0.f;

    const int arow = tid >> 1, ahalf = tid & 1;       // A: 128 rows x 32 cols
    const int bkr  = tid >> 3, bnc = (tid & 7) << 4;  // B: 32 rows x 128 cols

    for (int k0 = 0; k0 < K; k0 += BK) {
        // A tile: straight copy -> async global->LDS path
        const unsigned short* ap = A + (size_t)(bm + arow) * K + k0 + ahalf * 16;
        unsigned short* ad = &As[arow][ahalf * 16];
        cp16(ad, ap);
        cp16(ad + 8, ap + 8);
        // B tile: needs register transpose -> plain loads + ds stores
        const unsigned short* bp = B + (size_t)(k0 + bkr) * N + bn + bnc;
        uint4 b0 = *(const uint4*)bp;
        uint4 b1 = *(const uint4*)(bp + 8);
        unsigned short tmp[16];
        *(uint4*)(tmp)     = b0;
        *(uint4*)(tmp + 8) = b1;
#pragma unroll
        for (int i = 0; i < 16; ++i) Bt[bnc + i][bkr] = tmp[i];  // transpose
        cp_fence();          // drain ASYNCcnt before making LDS visible
        __syncthreads();

        // A fragment: lane lm holds row M=lm, k-chunks at lg*8 and lg*8+16
        v16bf af[4], bfv[2];
#pragma unroll
        for (int mi = 0; mi < 4; ++mi) {
            const unsigned short* r = &As[wm * 64 + mi * 16 + lm][lg * 8];
            af[mi] = ldfrag(r, r + 16);
        }
        // B fragment: lane lm holds col N=lm, k-chunks at lg*8 and lg*8+16 in Bt
#pragma unroll
        for (int ni = 0; ni < 2; ++ni) {
            const unsigned short* r = &Bt[wn * 32 + ni * 16 + lm][lg * 8];
            bfv[ni] = ldfrag(r, r + 16);
        }
#pragma unroll
        for (int mi = 0; mi < 4; ++mi)
#pragma unroll
            for (int ni = 0; ni < 2; ++ni)
                acc[mi][ni].v = __builtin_amdgcn_wmma_f32_16x16x32_bf16(
                    false, af[mi], false, bfv[ni], (short)0, acc[mi][ni].v,
                    false, false);
        __syncthreads();
    }

    // C layout: VGPR r -> row r + 8*lg; lane -> col lm
#pragma unroll
    for (int mi = 0; mi < 4; ++mi)
#pragma unroll
        for (int ni = 0; ni < 2; ++ni) {
            const int n = bn + wn * 32 + ni * 16 + lm;
            const float bv = bias[n];
#pragma unroll
            for (int r = 0; r < 8; ++r) {
                const int m = bm + wm * 64 + mi * 16 + r + lg * 8;
                store_out(&C[(size_t)m * N + n], acc[mi][ni].f[r] + bv);
            }
        }
}

// ---------- flash attention (causal, bf16 WMMA, fp32 softmax) ----------
// Grid: (T/64, B*H). Block: 128 threads = 4 waves; wave w owns q rows w*16..w*16+15.
__global__ __launch_bounds__(128)
void flash_attn_wmma(const unsigned short* __restrict__ qkv,  // [B*T][3C] bf16
                     unsigned short* __restrict__ O) {        // [B*T][C] bf16
    constexpr int T = 2048, C = 1024, C3 = 3072, PAD = 8;
    __shared__ unsigned short Qs[64][64 + PAD];   // [q][d]
    __shared__ unsigned short Ks[64][64 + PAD];   // [k][d]  (= B layout for Q*K^T)
    __shared__ unsigned short Vt[64][64 + PAD];   // [d][k]  (= B layout for P*V)
    __shared__ unsigned short Ps[4][16][64 + PAD];// per-wave P re-layout patch

    const int qt  = blockIdx.x;
    const int b   = blockIdx.y >> 4, h = blockIdx.y & 15;
    const int tid = threadIdx.x, wave = tid >> 5, lane = tid & 31;
    const int lm  = lane & 15, lg = lane >> 4;
    const int q0  = qt * 64;
    const unsigned short* base = qkv + (size_t)b * T * C3;

    const int row = tid >> 1, half = tid & 1;   // 2 threads per 64-elem row

    // ---- Q tile: TDM DMA if available, else (async) per-thread copy ----
#ifdef HAVE_TDM
    if (wave == 0)
        tdm_load_64x64_bf16((unsigned)(size_t)&Qs[0][0],
                            base + (size_t)q0 * C3 + h * 64, C3);
#else
    {
        const unsigned short* src = base + (size_t)(q0 + row) * C3 + h * 64 + half * 32;
        unsigned short* dst = &Qs[row][half * 32];
        cp16(dst,      src);
        cp16(dst +  8, src + 8);
        cp16(dst + 16, src + 16);
        cp16(dst + 24, src + 24);
    }
#endif

    F8 o[4];
    float mrow[8], lrow[8];
#pragma unroll
    for (int di = 0; di < 4; ++di)
#pragma unroll
        for (int i = 0; i < 8; ++i) o[di].f[i] = 0.f;
#pragma unroll
    for (int r = 0; r < 8; ++r) { mrow[r] = -INFINITY; lrow[r] = 0.f; }

    for (int j = 0; j <= qt; ++j) {
        const int kb = j * 64;
        // ---- K tile ----
#ifdef HAVE_TDM
        if (wave == 0)
            tdm_load_64x64_bf16((unsigned)(size_t)&Ks[0][0],
                                base + (size_t)kb * C3 + C + h * 64, C3);
#else
        {
            const unsigned short* ks = base + (size_t)(kb + row) * C3 + C + h * 64 + half * 32;
            unsigned short* kd = &Ks[row][half * 32];
            cp16(kd,      ks);
            cp16(kd +  8, ks + 8);
            cp16(kd + 16, ks + 16);
            cp16(kd + 24, ks + 24);
        }
#endif
        // ---- V tile, transposed into LDS (register transpose) ----
        {
            const unsigned short* vs = base + (size_t)(kb + row) * C3 + 2 * C + h * 64 + half * 32;
            unsigned short tmp[32];
            *(uint4*)(tmp)      = *(const uint4*)(vs);
            *(uint4*)(tmp +  8) = *(const uint4*)(vs + 8);
            *(uint4*)(tmp + 16) = *(const uint4*)(vs + 16);
            *(uint4*)(tmp + 24) = *(const uint4*)(vs + 24);
#pragma unroll
            for (int i = 0; i < 32; ++i) Vt[half * 32 + i][row] = tmp[i];
        }
#ifdef HAVE_TDM
        if (wave == 0) __builtin_amdgcn_s_wait_tensorcnt(0);
#else
        cp_fence();
#endif
        __syncthreads();

        // ---- S = Q * K^T (d = 64 -> two 32-wide wmma steps) ----
        F8 s[4];
#pragma unroll
        for (int ni = 0; ni < 4; ++ni)
#pragma unroll
            for (int i = 0; i < 8; ++i) s[ni].f[i] = 0.f;
#pragma unroll
        for (int ks = 0; ks < 2; ++ks) {
            const unsigned short* qr = &Qs[wave * 16 + lm][ks * 32 + lg * 8];
            const v16bf qf = ldfrag(qr, qr + 16);
            v16bf kf[4];
#pragma unroll
            for (int ni = 0; ni < 4; ++ni) {
                const unsigned short* r = &Ks[ni * 16 + lm][ks * 32 + lg * 8];
                kf[ni] = ldfrag(r, r + 16);
            }
#pragma unroll
            for (int ni = 0; ni < 4; ++ni)
                s[ni].v = __builtin_amdgcn_wmma_f32_16x16x32_bf16(
                    false, qf, false, kf[ni], (short)0, s[ni].v, false, false);
        }

        // ---- scale + causal mask + online softmax ----
        const float scale = 0.125f;   // 1/sqrt(64)
        float mnew[8];
#pragma unroll
        for (int r = 0; r < 8; ++r) {
            float mx = -INFINITY;
            const int qg = q0 + wave * 16 + r + lg * 8;
#pragma unroll
            for (int ni = 0; ni < 4; ++ni) {
                float x = s[ni].f[r] * scale;
                const int kg = kb + ni * 16 + lm;
                x = (kg > qg) ? -INFINITY : x;
                s[ni].f[r] = x;
                mx = fmaxf(mx, x);
            }
            mx = fmaxf(mx, __shfl_xor(mx, 1, 32));
            mx = fmaxf(mx, __shfl_xor(mx, 2, 32));
            mx = fmaxf(mx, __shfl_xor(mx, 4, 32));
            mx = fmaxf(mx, __shfl_xor(mx, 8, 32));
            mnew[r] = fmaxf(mrow[r], mx);
        }
#pragma unroll
        for (int r = 0; r < 8; ++r) {
            const float corr = __expf(mrow[r] - mnew[r]);
            mrow[r] = mnew[r];
            lrow[r] *= corr;
#pragma unroll
            for (int di = 0; di < 4; ++di) o[di].f[r] *= corr;
        }
#pragma unroll
        for (int r = 0; r < 8; ++r) {
            float rs = 0.f;
#pragma unroll
            for (int ni = 0; ni < 4; ++ni) {
                const float p = __expf(s[ni].f[r] - mnew[r]);
                s[ni].f[r] = p;
                rs += p;
            }
            rs += __shfl_xor(rs, 1, 32);
            rs += __shfl_xor(rs, 2, 32);
            rs += __shfl_xor(rs, 4, 32);
            rs += __shfl_xor(rs, 8, 32);
            lrow[r] += rs;
        }

        // ---- P: C-layout -> A-layout via per-wave LDS patch (same-wave LDS in order) ----
#pragma unroll
        for (int ni = 0; ni < 4; ++ni)
#pragma unroll
            for (int r = 0; r < 8; ++r)
                Ps[wave][r + lg * 8][ni * 16 + lm] = f2bf(s[ni].f[r]);

        // ---- O += P * V (Vt[d][k] is exactly the B fragment layout) ----
#pragma unroll
        for (int ks = 0; ks < 2; ++ks) {
            const unsigned short* pr = &Ps[wave][lm][ks * 32 + lg * 8];
            const v16bf pf = ldfrag(pr, pr + 16);
            v16bf vf[4];
#pragma unroll
            for (int di = 0; di < 4; ++di) {
                const unsigned short* r = &Vt[di * 16 + lm][ks * 32 + lg * 8];
                vf[di] = ldfrag(r, r + 16);
            }
#pragma unroll
            for (int di = 0; di < 4; ++di)
                o[di].v = __builtin_amdgcn_wmma_f32_16x16x32_bf16(
                    false, pf, false, vf[di], (short)0, o[di].v, false, false);
        }
        __syncthreads();   // before next tile overwrites Ks/Vt
    }

    // ---- normalize + store O as bf16 [b*T+t][h*64+d] ----
#pragma unroll
    for (int r = 0; r < 8; ++r) {
        const float inv = 1.0f / lrow[r];
        const int trow = q0 + wave * 16 + r + lg * 8;
        unsigned short* dst = O + (size_t)(b * T + trow) * C + h * 64;
#pragma unroll
        for (int di = 0; di < 4; ++di)
            dst[di * 16 + lm] = f2bf(o[di].f[r] * inv);
    }
}

// ---------- host launch ----------
extern "C" void kernel_launch(void* const* d_in, const int* in_sizes, int n_in,
                              void* d_out, int out_size, void* d_ws, size_t ws_size,
                              hipStream_t stream) {
    const float* X  = (const float*)d_in[0];   // [2,2048,1024]
    const float* Wa = (const float*)d_in[1];   // [1024,3072]
    const float* ba = (const float*)d_in[2];   // [3072]
    const float* Wp = (const float*)d_in[3];   // [1024,1024]
    const float* bp = (const float*)d_in[4];   // [1024]
    float* out = (float*)d_out;                // [2,2048,1024] fp32

    const int Bb = 2, T = 2048, C = 1024, M = Bb * T, N3 = 3 * C;

    // bf16 staging in workspace (~48 MB total)
    unsigned short* Xbf  = (unsigned short*)d_ws;
    unsigned short* Wabf = Xbf  + (size_t)M * C;
    unsigned short* Wpbf = Wabf + (size_t)C * N3;
    unsigned short* QKV  = Wpbf + (size_t)C * C;
    unsigned short* Obf  = QKV  + (size_t)M * N3;

    int n;
    n = M * C;
    f32_to_bf16_kernel<<<(n / 4 + 255) / 256, 256, 0, stream>>>(X, Xbf, n);
    n = C * N3;
    f32_to_bf16_kernel<<<(n / 4 + 255) / 256, 256, 0, stream>>>(Wa, Wabf, n);
    n = C * C;
    f32_to_bf16_kernel<<<(n / 4 + 255) / 256, 256, 0, stream>>>(Wp, Wpbf, n);

    // QKV = X * W_attn + b_attn  -> bf16
    gemm_bf16_wmma<unsigned short>
        <<<dim3(N3 / 128, M / 128), 256, 0, stream>>>(Xbf, Wabf, ba, QKV, M, N3, C);

    // causal flash attention per (b,h,64-query tile)
    flash_attn_wmma<<<dim3(T / 64, Bb * 16), 128, 0, stream>>>(QKV, Obf);

    // out = O * W_proj + b_proj -> fp32
    gemm_bf16_wmma<float>
        <<<dim3(C / 128, M / 128), 256, 0, stream>>>(Obf, Wpbf, bp, out, M, C, C);
}
